// OpenCUA_VisionAttention_23330262352197
// MI455X (gfx1250) — matrix operand order repaired and checked by reading, not compiled, not run on hardware
//
#include <hip/hip_runtime.h>
#include <hip/hip_bf16.h>

typedef __attribute__((ext_vector_type(16))) _Float16 v16h;
typedef __attribute__((ext_vector_type(8)))  _Float16 v8h;
typedef __attribute__((ext_vector_type(8)))  float    v8f;
typedef int v4i_ __attribute__((vector_size(16)));

#define AS1 __attribute__((address_space(1)))
#define AS3 __attribute__((address_space(3)))

#if __has_builtin(__builtin_amdgcn_global_load_async_to_lds_b128)
#define ASYNC_LDS 1
#else
#define ASYNC_LDS 0
#endif

// ---------------------------------------------------------------------------
// 16B global -> LDS copy. Async (ASYNCcnt, no VGPR round trip) when available.
// ---------------------------------------------------------------------------
__device__ inline void cp16_g2l(const _Float16* g, _Float16* l) {
#if ASYNC_LDS
  __builtin_amdgcn_global_load_async_to_lds_b128((AS1 v4i_*)g, (AS3 v4i_*)l, 0,
                                                 0);
#else
  *(v8h*)l = *(const v8h*)g;
#endif
}

__device__ inline void wait_async0() {
#if ASYNC_LDS
#if __has_builtin(__builtin_amdgcn_s_wait_asynccnt)
  __builtin_amdgcn_s_wait_asynccnt(0);
#else
  asm volatile("s_wait_asynccnt 0" ::: "memory");
#endif
#endif
}

// ---------------------------------------------------------------------------
// WMMA helpers (CDNA5 gfx1250: V_WMMA_F32_16X16X32_F16, wave32)
// ---------------------------------------------------------------------------
__device__ inline v8f wmma_f16(v16h a, v16h b, v8f c) {
  return __builtin_amdgcn_wmma_f32_16x16x32_f16(false, a, false, b, (short)0, c,
                                                false, false);
}

// A-matrix fragment: 16x32 f16 (ISA 7.12.2 layout), row-major storage.
__device__ inline v16h load_frag_a(const _Float16* p, int stride) {
  const int lane = threadIdx.x & 31;
  const int m  = lane & 15;
  const int kb = (lane >> 4) << 3;      // 0 or 8
  const v8h lo = *(const v8h*)(p + m * stride + kb);
  const v8h hi = *(const v8h*)(p + m * stride + kb + 16);
  return __builtin_shufflevector(lo, hi, 0, 1, 2, 3, 4, 5, 6, 7,
                                 8, 9, 10, 11, 12, 13, 14, 15);
}

// B-matrix fragment 32x16 f16 from n-major ([N,K] row-major) storage.
__device__ inline v16h load_frag_b_nk(const _Float16* p, int stride) {
  const int lane = threadIdx.x & 31;
  const int n  = lane & 15;
  const int kb = (lane >> 4) << 3;      // 0 or 8
  const v8h lo = *(const v8h*)(p + n * stride + kb);
  const v8h hi = *(const v8h*)(p + n * stride + kb + 16);
  return __builtin_shufflevector(lo, hi, 0, 1, 2, 3, 4, 5, 6, 7,
                                 8, 9, 10, 11, 12, 13, 14, 15);
}

// ---------------------------------------------------------------------------
// 16-lane-row reductions via DPP ROW_XMASK (pure VALU, no DS traffic).
// dpp_ctrl 0x160|mask: lane <- lane ^ mask within each row of 16.
// ---------------------------------------------------------------------------
template <int CTRL>
__device__ inline float dpp_xmask(float x) {
  return __builtin_bit_cast(
      float, __builtin_amdgcn_update_dpp(0, __builtin_bit_cast(int, x), CTRL,
                                         0xF, 0xF, true));
}
__device__ inline float row16_max(float x) {
  x = fmaxf(x, dpp_xmask<0x161>(x));
  x = fmaxf(x, dpp_xmask<0x162>(x));
  x = fmaxf(x, dpp_xmask<0x164>(x));
  x = fmaxf(x, dpp_xmask<0x168>(x));
  return x;
}
__device__ inline float row16_sum(float x) {
  x += dpp_xmask<0x161>(x);
  x += dpp_xmask<0x162>(x);
  x += dpp_xmask<0x164>(x);
  x += dpp_xmask<0x168>(x);
  return x;
}

// ---------------------------------------------------------------------------
// fp32 -> f16 conversion
// ---------------------------------------------------------------------------
__global__ void cvt_f32_f16(const float* __restrict__ in,
                            _Float16* __restrict__ out, int n) {
  for (int i = blockIdx.x * 256 + threadIdx.x; i < n; i += gridDim.x * 256)
    out[i] = (_Float16)in[i];
}

// ---------------------------------------------------------------------------
// Tiled WMMA GEMM:  C[M,N] = A[M,K] * W[N,K]^T + bias[N]
// Double-buffered LDS tiles filled by async global->LDS B128 copies:
//   issue(t+1) right after the barrier, compute(t) overlaps the DMA.
// Block: 256 threads (8 waves). Tile: 128(M) x 64(N), K-step 32.
// ---------------------------------------------------------------------------
template <bool F16OUT>
__global__ void __launch_bounds__(256)
gemm_wmma(const _Float16* __restrict__ A, const _Float16* __restrict__ W,
          const float* __restrict__ bias, void* __restrict__ Cout,
          int M, int N, int K) {
  constexpr int TM = 128, TN = 64, TK = 32;
  constexpr int ASTR = 40, BSTR = 40;  // padded strides (80B, 16B-aligned rows)
  __shared__ alignas(16) _Float16 As[2][TM * ASTR];
  __shared__ alignas(16) _Float16 Bs[2][TN * BSTR];

  const int tid  = threadIdx.x;
  const int wave = tid >> 5;
  const int m0   = blockIdx.y * TM;
  const int n0   = blockIdx.x * TN;

  auto issue = [&](int k0, int buf) {
    // A tile: 128 rows x 32 halves = 512 16B chunks (2/thread).
#pragma unroll
    for (int v = tid; v < (TM * TK) / 8; v += 256) {
      const int r = v >> 2, c = (v & 3) * 8;
      cp16_g2l(&A[(size_t)(m0 + r) * K + k0 + c], &As[buf][r * ASTR + c]);
    }
    // B tile: 64 rows x 32 halves = 256 chunks (1/thread).
    {
      const int r = tid >> 2, c = (tid & 3) * 8;
      cp16_g2l(&W[(size_t)(n0 + r) * K + k0 + c], &Bs[buf][r * BSTR + c]);
    }
  };

  v8f acc[4];
#pragma unroll
  for (int f = 0; f < 4; ++f)
#pragma unroll
    for (int j = 0; j < 8; ++j) acc[f][j] = 0.0f;

  const int NT = K / TK;
  issue(0, 0);
  for (int t = 0; t < NT; ++t) {
    wait_async0();        // own tile-t copies landed
    __syncthreads();      // everyone's tile-t landed; tile-(t+1) buffer free
    if (t + 1 < NT) issue((t + 1) * TK, (t + 1) & 1);

    const _Float16* as = As[t & 1];
    const _Float16* bs = Bs[t & 1];
    const v16h a = load_frag_a(&as[wave * 16 * ASTR], ASTR);
#pragma unroll
    for (int nf = 0; nf < 4; ++nf) {
      const v16h b = load_frag_b_nk(&bs[nf * 16 * BSTR], BSTR);
      acc[nf] = wmma_f16(a, b, acc[nf]);
    }
    __syncthreads();      // compute(t) done before buffer t&1 is re-issued
  }

  const int lane = tid & 31;
  const int g = lane >> 4, n = lane & 15;
#pragma unroll
  for (int nf = 0; nf < 4; ++nf) {
    const int col = n0 + nf * 16 + n;
    const float b = bias[col];
#pragma unroll
    for (int j = 0; j < 8; ++j) {
      const int row = m0 + wave * 16 + g * 8 + j;
      const float val = acc[nf][j] + b;
      if (F16OUT)
        ((_Float16*)Cout)[(size_t)row * N + col] = (_Float16)val;
      else
        ((float*)Cout)[(size_t)row * N + col] = val;
    }
  }
}

// ---------------------------------------------------------------------------
// Vision RoPE, in place on the q and k thirds of qkv (f16).
// ---------------------------------------------------------------------------
__global__ void rope_kernel(_Float16* __restrict__ qkv,
                            const float* __restrict__ rpe) {
  const int idx = blockIdx.x * 256 + threadIdx.x;
  if (idx >= 8192 * 16 * 40) return;
  const int i = idx % 40;
  const int h = (idx / 40) % 16;
  const int s = idx / (40 * 16);
  const float th = rpe[s * 40 + i];
  const float c = __cosf(th), sn = __sinf(th);
  const size_t base = (size_t)s * 3840 + h * 80;
#pragma unroll
  for (int which = 0; which < 2; ++which) {  // 0 = q, 1 = k
    _Float16* p = qkv + base + which * 1280;
    const float x1 = (float)p[i], x2 = (float)p[i + 40];
    p[i]      = (_Float16)(x1 * c - x2 * sn);
    p[i + 40] = (_Float16)(x2 * c + x1 * sn);
  }
}

// ---------------------------------------------------------------------------
// Flash attention over block-diagonal segments.
// Grid: (qblock=8, head=16, seg=8). Block: 256 threads = 8 waves; each wave
// owns a 16-row q strip. Head dim 80 padded to 96 for QK^T. K tiles stream
// through LDS via async B128 copies; V tiles are transposed on the fly.
// ---------------------------------------------------------------------------
__global__ void __launch_bounds__(256)
attn_kernel(const _Float16* __restrict__ qkv, _Float16* __restrict__ ctx) {
  constexpr int L = 1024, KT = 32, KSTR = 96;
  __shared__ alignas(16) _Float16 kt[32 * KSTR];    // [key][d], d padded to 96
  __shared__ alignas(16) _Float16 vtT[80 * 32];     // [d][key] (transposed)
  __shared__ alignas(16) _Float16 pws[8 * 16 * 32]; // per-wave P scratch

  const int tid = threadIdx.x, lane = tid & 31, wave = tid >> 5;
  const int qb = blockIdx.x, head = blockIdx.y, seg = blockIdx.z;
  const int srow0 = seg * L + qb * 128 + wave * 16;
  const size_t qcol = (size_t)head * 80;

  // Zero the static pad region kt[key][80..96) once.
  for (int v = tid; v < 32 * 16; v += 256)
    kt[(v >> 4) * KSTR + 80 + (v & 15)] = (_Float16)0.0f;

  // Q A-fragments for this wave's 16 rows, d padded with zeros to 96.
  v16h qa[3];
  {
    const int m = lane & 15, kb = (lane >> 4) << 3;
    const _Float16* qrow = qkv + (size_t)(srow0 + m) * 3840 + qcol;
#pragma unroll
    for (int t = 0; t < 3; ++t) {
      v16h r;
#pragma unroll
      for (int hh = 0; hh < 16; ++hh) {
        const int d = t * 32 + kb + (hh & 7) + ((hh >> 3) << 4);
        r[hh] = (d < 80) ? qrow[d] : (_Float16)0.0f;
      }
      qa[t] = r;
    }
  }

  v8f o[5];
  float mrun[8], lrun[8];
#pragma unroll
  for (int f = 0; f < 5; ++f)
#pragma unroll
    for (int j = 0; j < 8; ++j) o[f][j] = 0.0f;
#pragma unroll
  for (int j = 0; j < 8; ++j) { mrun[j] = -1e30f; lrun[j] = 0.0f; }

  const float scale = 0.111803398874989485f;  // 1/sqrt(80)
  const size_t kbase = (size_t)seg * L * 3840 + 1280 + qcol;
  const size_t vbase = (size_t)seg * L * 3840 + 2560 + qcol;
  _Float16* mypws = pws + wave * 16 * 32;

  for (int t0 = 0; t0 < L; t0 += KT) {
    __syncthreads();
    // K tile: 32 rows x 80 halves = 320 16B chunks, async to LDS.
#pragma unroll
    for (int v = tid; v < 32 * 10; v += 256) {
      const int key = v / 10, c = (v % 10) * 8;
      cp16_g2l(&qkv[kbase + (size_t)(t0 + key) * 3840 + c],
               &kt[key * KSTR + c]);
    }
    // V tile transposed -> vtT[d][key] (contiguous B-fragments for PV).
    for (int idx = tid; idx < 32 * 80; idx += 256) {
      const int key = idx / 80, d = idx % 80;
      vtT[d * 32 + key] = qkv[vbase + (size_t)(t0 + key) * 3840 + d];
    }
    wait_async0();
    __syncthreads();

    // S = Q K^T for 32 keys: two 16x16 C fragments, 3 K-steps of 32 over d.
    v8f s0, s1;
#pragma unroll
    for (int j = 0; j < 8; ++j) { s0[j] = 0.0f; s1[j] = 0.0f; }
#pragma unroll
    for (int t = 0; t < 3; ++t) {
      const v16h b0 = load_frag_b_nk(&kt[t * 32], KSTR);             // keys 0-15
      const v16h b1 = load_frag_b_nk(&kt[16 * KSTR + t * 32], KSTR); // keys 16-31
      s0 = wmma_f16(qa[t], b0, s0);
      s1 = wmma_f16(qa[t], b1, s1);
    }

    // Online softmax; row stats reduced across the 16-lane half-group with
    // DPP ROW_XMASK (VALU-only, keeps the DS port free for fragment loads).
    float p0[8], p1[8];
#pragma unroll
    for (int j = 0; j < 8; ++j) {
      const float a = s0[j] * scale, b = s1[j] * scale;
      const float mx = row16_max(fmaxf(a, b));
      const float mnew = fmaxf(mrun[j], mx);
      const float corr = __expf(mrun[j] - mnew);
      const float e0 = __expf(a - mnew), e1 = __expf(b - mnew);
      const float rs = row16_sum(e0 + e1);
      lrun[j] = lrun[j] * corr + rs;
      mrun[j] = mnew;
      p0[j] = e0; p1[j] = e1;
#pragma unroll
      for (int f = 0; f < 5; ++f) o[f][j] *= corr;
    }

    // C-layout P -> A-layout via per-wave LDS round trip (wave-local).
    {
      const int g = lane >> 4, n = lane & 15;
#pragma unroll
      for (int j = 0; j < 8; ++j) {
        mypws[(j + 8 * g) * 32 + n]      = (_Float16)p0[j];
        mypws[(j + 8 * g) * 32 + 16 + n] = (_Float16)p1[j];
      }
    }
    asm volatile("s_wait_dscnt 0" ::: "memory");
    const v16h pa = load_frag_a(mypws, 32);

    // O += P * V  (K-dim = 32 keys, N-dim = 80 in 5 chunks of 16).
#pragma unroll
    for (int f = 0; f < 5; ++f) {
      const v16h bv = load_frag_b_nk(&vtT[f * 16 * 32], 32);
      o[f] = wmma_f16(pa, bv, o[f]);
    }
  }

  // Normalize and write context (f16).
  {
    const int g = lane >> 4, n = lane & 15;
#pragma unroll
    for (int f = 0; f < 5; ++f)
#pragma unroll
      for (int j = 0; j < 8; ++j) {
        const int row = srow0 + j + 8 * g;
        ctx[(size_t)row * 1280 + head * 80 + f * 16 + n] =
            (_Float16)(o[f][j] / lrun[j]);
      }
  }
}

// ---------------------------------------------------------------------------
// Launch pipeline.
// ---------------------------------------------------------------------------
extern "C" void kernel_launch(void* const* d_in, const int* in_sizes, int n_in,
                              void* d_out, int out_size, void* d_ws,
                              size_t ws_size, hipStream_t stream) {
  (void)in_sizes; (void)n_in; (void)out_size; (void)ws_size;
  const float* x      = (const float*)d_in[0];
  // d_in[1] = cu_seqlens (equal 1024-length segments; folded into grid dims)
  const float* rpe    = (const float*)d_in[2];
  const float* qkv_w  = (const float*)d_in[3];
  const float* qkv_b  = (const float*)d_in[4];
  const float* proj_w = (const float*)d_in[5];
  const float* proj_b = (const float*)d_in[6];
  float* out = (float*)d_out;

  char* ws = (char*)d_ws;
  _Float16* xh    = (_Float16*)(ws);                    // 20,971,520 B
  _Float16* wqkvh = (_Float16*)(ws + 20971520);         //  9,830,400 B
  _Float16* wph   = (_Float16*)(ws + 30801920);         //  3,276,800 B
  _Float16* qkvh  = (_Float16*)(ws + 34078720);         // 62,914,560 B
  _Float16* ctxh  = (_Float16*)(ws + 96993280);         // 20,971,520 B

  cvt_f32_f16<<<4096, 256, 0, stream>>>(x, xh, 8192 * 1280);
  cvt_f32_f16<<<4096, 256, 0, stream>>>(qkv_w, wqkvh, 3840 * 1280);
  cvt_f32_f16<<<2048, 256, 0, stream>>>(proj_w, wph, 1280 * 1280);

  gemm_wmma<true><<<dim3(3840 / 64, 8192 / 128), 256, 0, stream>>>(
      xh, wqkvh, qkv_b, (void*)qkvh, 8192, 3840, 1280);

  rope_kernel<<<(8192 * 16 * 40 + 255) / 256, 256, 0, stream>>>(qkvh, rpe);

  attn_kernel<<<dim3(8, 16, 8), 256, 0, stream>>>(qkvh, ctxh);

  gemm_wmma<false><<<dim3(1280 / 64, 8192 / 128), 256, 0, stream>>>(
      ctxh, wph, proj_b, (void*)out, 8192, 1280, 1280);
}
